// TransformerEncoderLayer_283467841809
// MI455X (gfx1250) — matrix operand order repaired and checked by reading, not compiled, not run on hardware
//
#include <hip/hip_runtime.h>
#include <hip/hip_bf16.h>
#include <math.h>

// ---------------------------------------------------------------------------
// MI455X (gfx1250) transformer encoder layer.
// All heavy GEMMs run on v_wmma_f32_16x16x32_bf16 (wave32, 16x16x32 tiles).
// LDS staging uses the Tensor Data Mover / async-to-LDS path when the
// toolchain exposes it (guarded; falls back to sync loads).
// ---------------------------------------------------------------------------

typedef __attribute__((ext_vector_type(16))) __bf16   v16bf;
typedef __attribute__((ext_vector_type(8)))  float    v8f;
typedef __attribute__((ext_vector_type(4)))  unsigned u32x4;
typedef __attribute__((ext_vector_type(4)))  float    f32x4;

union AB16 { v16bf v; u32x4 u[2]; };

static __device__ __forceinline__ v8f wmma_bf16(const AB16& a, const AB16& b, v8f c) {
  return __builtin_amdgcn_wmma_f32_16x16x32_bf16(false, a.v, false, b.v,
                                                 (short)0, c, false, false);
}

#define NT_   8192   // N*S tokens
#define D_    1024
#define H_    16
#define HD_   64
#define S_    1024
#define FFN_  4096

// ------------------- CDNA5 async / TDM data-movement paths -----------------
#if defined(__AMDGCN__) && __has_builtin(__builtin_amdgcn_tensor_load_to_lds) && \
    __has_builtin(__builtin_amdgcn_s_wait_tensorcnt)
#define HAVE_TDM 1
#endif
#if defined(__AMDGCN__) && __has_builtin(__builtin_amdgcn_global_load_async_to_lds_b128) && \
    __has_builtin(__builtin_amdgcn_s_wait_asynccnt)
#define HAVE_ASYNC 1
#endif

#if defined(HAVE_ASYNC)
// Builtin signature (from hipcc diagnostic): (v4i AS1*, v4i AS3*, imm, imm).
typedef __attribute__((__vector_size__(16))) int vi4;
typedef __attribute__((address_space(1))) vi4 g_vi4;
typedef __attribute__((address_space(3))) vi4 l_vi4;
// Flat LDS addresses carry the LDS byte offset in addr[31:0] (ISA 10.2).
static __device__ __forceinline__ void async_b128(const void* g, const void* l) {
  __builtin_amdgcn_global_load_async_to_lds_b128(
      (g_vi4*)(unsigned long long)(uintptr_t)g,
      (l_vi4*)(unsigned)(uintptr_t)l, 0, 0);
}
#endif

#if defined(HAVE_TDM)
typedef __attribute__((ext_vector_type(4))) unsigned tdm_u4;
typedef __attribute__((ext_vector_type(8))) int      tdm_i8;
typedef __attribute__((ext_vector_type(4))) int      tdm_i4;

// 2D bf16 tile DMA: tile_dim0=32 elems (64B) x tile_dim1=128 rows, row stride
// in LDS padded to 80B via pad_interval=16 DWORDs, pad_amount=4 DWORDs.
static __device__ __forceinline__ void tdm_load_tile_128x32(
    const void* gptr, const void* lptr, unsigned K_elems) {
  unsigned long long ga = (unsigned long long)(uintptr_t)gptr;
  tdm_u4 g0 = (tdm_u4)0;
  g0[0] = 1u;                                             // count=1 (user D#)
  g0[1] = (unsigned)(uintptr_t)lptr;                      // lds_addr
  g0[2] = (unsigned)(ga & 0xFFFFFFFFull);                 // global_addr[31:0]
  g0[3] = (unsigned)((ga >> 32) & 0x01FFFFFFull) | 0x80000000u; // [56:32]|type=2
  tdm_i8 g1 = (tdm_i8)0;
  g1[0] = (int)((1u << 16) |            // data_size = 2B
                (1u << 20) |            // pad_enable
                (3u << 22) |            // pad_interval: 16 DWORDs (one row)
                (3u << 25));            // pad_amount: 4 DWORDs (16B)
  g1[1] = (int)((K_elems & 0xFFFFu) << 16);               // tensor_dim0[15:0]
  g1[2] = (int)(((K_elems >> 16) & 0xFFFFu) | (128u << 16)); // dim0 hi | dim1
  g1[3] = (int)(32u << 16);                               // tile_dim0 = 32
  g1[4] = (int)128u;                                      // tile_dim1 = 128
  g1[5] = (int)K_elems;                                   // tensor_dim0_stride
  tdm_i4 z4 = (tdm_i4)0;
#if __clang_major__ >= 23
  tdm_i8 z8 = (tdm_i8)0;
  __builtin_amdgcn_tensor_load_to_lds(g0, g1, z4, z4, z8, 0);
#else
  __builtin_amdgcn_tensor_load_to_lds(g0, g1, z4, z4, 0);
#endif
}
#endif  // HAVE_TDM

// -------------------------- elementwise converts ---------------------------
__global__ __launch_bounds__(256) void f2bf_kernel(const float* __restrict__ in,
                                                   __bf16* __restrict__ out) {
  int i = (blockIdx.x * 256 + threadIdx.x) * 4;
  f32x4 v = *(const f32x4*)(in + i);
  out[i + 0] = (__bf16)v.x; out[i + 1] = (__bf16)v.y;
  out[i + 2] = (__bf16)v.z; out[i + 3] = (__bf16)v.w;
}

// W [K][N] fp32 -> Wt [N][K] bf16 (so B fragments read contiguously along K)
__global__ __launch_bounds__(256) void wconvert_kernel(const float* __restrict__ W,
                                                       __bf16* __restrict__ Wt,
                                                       int K, int N) {
  int idx = blockIdx.x * 256 + threadIdx.x;      // over N*K outputs
  int n = idx / K, k = idx - n * K;
  Wt[idx] = (__bf16)W[(size_t)k * N + n];
}

// ------------------------------- WMMA GEMM ---------------------------------
// C[M][N] = act( (A[M][K](bf16) x Bt[N][K](bf16)^T + bias) * scale )
// MODE 0: f32 out.  MODE 1: exact-GELU, bf16 out.  MODE 2: bf16 head-split QKV.
template <int MODE>
__global__ __launch_bounds__(256)
void gemm_bf16_kernel(const __bf16* __restrict__ A, const __bf16* __restrict__ Bt,
                      const float* __restrict__ bias, void* __restrict__ outp,
                      int M, int N, int K, float scale) {
  __shared__ __bf16 As[128][40];   // +8 pad: 80B row stride, conflict-free frags
  __shared__ __bf16 Bs[128][40];

  const int tid  = threadIdx.x;
  const int lane = tid & 31, w = tid >> 5;
  const int wm = w >> 1, wn = w & 1;          // 4 x 2 wave grid
  const int hi = lane >> 4, ln = lane & 15;
  const size_t rowBase = (size_t)blockIdx.y * 128;
  const size_t colBase = (size_t)blockIdx.x * 128;

  const int lrow = tid >> 1;                  // 0..127
  const int lseg = (tid & 1) * 16;            // 0 / 16

  v8f acc[2][4] = {};

  for (int k0 = 0; k0 < K; k0 += 32) {
    __syncthreads();
#if defined(HAVE_TDM)
    if (tid == 0) {                          // one TDM DMA per 8KB tile
      tdm_load_tile_128x32(A  + rowBase * (size_t)K + k0, &As[0][0], (unsigned)K);
      tdm_load_tile_128x32(Bt + colBase * (size_t)K + k0, &Bs[0][0], (unsigned)K);
      __builtin_amdgcn_s_wait_tensorcnt(0);
    }
#else
    {
      const __bf16* srcA = A  + (rowBase + lrow) * (size_t)K + k0 + lseg;
      const __bf16* srcB = Bt + (colBase + lrow) * (size_t)K + k0 + lseg;
#if defined(HAVE_ASYNC)
      async_b128(srcA,     &As[lrow][lseg]);
      async_b128(srcA + 8, &As[lrow][lseg + 8]);
      async_b128(srcB,     &Bs[lrow][lseg]);
      async_b128(srcB + 8, &Bs[lrow][lseg + 8]);
      __builtin_amdgcn_s_wait_asynccnt(0);
#else
      if (k0 + 32 < K) {
        __builtin_prefetch((const void*)(srcA + 32), 0, 1);
        __builtin_prefetch((const void*)(srcB + 32), 0, 1);
      }
      *(u32x4*)(&As[lrow][lseg])     = *(const u32x4*)(srcA);
      *(u32x4*)(&As[lrow][lseg + 8]) = *(const u32x4*)(srcA + 8);
      *(u32x4*)(&Bs[lrow][lseg])     = *(const u32x4*)(srcB);
      *(u32x4*)(&Bs[lrow][lseg + 8]) = *(const u32x4*)(srcB + 8);
#endif
    }
#endif
    __syncthreads();

    AB16 a[2], b[4];
#pragma unroll
    for (int g = 0; g < 2; ++g) {             // A frag: row = lane&15, k split by hi
      const __bf16* p = &As[wm * 32 + g * 16 + ln][8 * hi];
      a[g].u[0] = *(const u32x4*)(p);
      a[g].u[1] = *(const u32x4*)(p + 16);
    }
#pragma unroll
    for (int c = 0; c < 4; ++c) {             // B frag: col = lane&15, k split by hi
      const __bf16* p = &Bs[wn * 64 + c * 16 + ln][8 * hi];
      b[c].u[0] = *(const u32x4*)(p);
      b[c].u[1] = *(const u32x4*)(p + 16);
    }
#pragma unroll
    for (int g = 0; g < 2; ++g)
#pragma unroll
      for (int c = 0; c < 4; ++c)
        acc[g][c] = wmma_bf16(a[g], b[c], acc[g][c]);
  }

#pragma unroll
  for (int g = 0; g < 2; ++g)
#pragma unroll
    for (int c = 0; c < 4; ++c) {
      int gcol = (int)colBase + wn * 64 + c * 16 + ln;
      float bv = bias[gcol];
#pragma unroll
      for (int r = 0; r < 8; ++r) {
        size_t grow = rowBase + wm * 32 + g * 16 + r + 8 * hi;   // C: M=r+8*hi
        float vv = (acc[g][c][r] + bv) * scale;
        if constexpr (MODE == 0) {
          ((float*)outp)[grow * (size_t)N + gcol] = vv;
        } else if constexpr (MODE == 1) {
          float ge = 0.5f * vv * (1.0f + erff(vv * 0.70710678118654752f));
          ((__bf16*)outp)[grow * (size_t)N + gcol] = (__bf16)ge;
        } else {                               // QKV -> [n*H+h][s][64]
          int n = (int)(grow >> 10), s = (int)(grow & 1023);
          int hh = gcol >> 6, e = gcol & 63;
          size_t oidx = (((size_t)(n * H_ + hh)) * S_ + s) * HD_ + e;
          ((__bf16*)outp)[oidx] = (__bf16)vv;
        }
      }
    }
}

// ---------------------------- flash attention ------------------------------
// q,k,v: [N*H][S][64] bf16 (q pre-scaled by 1/8). ctx out: token-major bf16.
__global__ __launch_bounds__(256)
void attn_kernel(const __bf16* __restrict__ q, const __bf16* __restrict__ k,
                 const __bf16* __restrict__ v, __bf16* __restrict__ ctx) {
  __shared__ __bf16 Ks[32][72];      // [key][hd], padded
  __shared__ __bf16 VTs[64][40];     // [hd][key], padded
  __shared__ __bf16 Ps[8][16 * 40];  // per-wave P tile (C->A layout bounce)

  const int tid = threadIdx.x, lane = tid & 31, w = tid >> 5;
  const int hi = lane >> 4, ln = lane & 15;
  const int nh = blockIdx.x, n = nh >> 4, h = nh & 15;
  const int qbase = blockIdx.y * 128 + w * 16;
  const size_t base = (size_t)nh * S_ * HD_;

  AB16 qa[2];
  {
    const __bf16* qp = q + base + (size_t)(qbase + ln) * HD_;
    qa[0].u[0] = *(const u32x4*)(qp + 8 * hi);
    qa[0].u[1] = *(const u32x4*)(qp + 16 + 8 * hi);
    qa[1].u[0] = *(const u32x4*)(qp + 32 + 8 * hi);
    qa[1].u[1] = *(const u32x4*)(qp + 48 + 8 * hi);
  }

  float mrow[8], lrow[8], alpha[8];
#pragma unroll
  for (int r = 0; r < 8; ++r) { mrow[r] = -INFINITY; lrow[r] = 0.f; }
  v8f co[4] = {};

  const int ldkey = tid >> 3;            // 0..31
  const int ldseg = (tid & 7) * 8;       // 0..56

  for (int kb = 0; kb < S_; kb += 32) {
    __syncthreads();
    {
      const __bf16* kp = k + base + (size_t)(kb + ldkey) * HD_ + ldseg;
#if defined(HAVE_ASYNC)
      async_b128(kp, &Ks[ldkey][ldseg]);
#else
      *(u32x4*)(&Ks[ldkey][ldseg]) = *(const u32x4*)kp;
#endif
      u32x4 uv = *(const u32x4*)(v + base + (size_t)(kb + ldkey) * HD_ + ldseg);
      const __bf16* pv = (const __bf16*)&uv;
#pragma unroll
      for (int i = 0; i < 8; ++i) VTs[ldseg + i][ldkey] = pv[i];   // transpose
#if defined(HAVE_ASYNC)
      __builtin_amdgcn_s_wait_asynccnt(0);
#endif
    }
    __syncthreads();

    // scores: s0 = keys [kb,kb+16), s1 = keys [kb+16,kb+32)
    v8f s0 = {}, s1 = {};
    AB16 bf;
    const __bf16* p;
    p = &Ks[ln][8 * hi];       bf.u[0] = *(const u32x4*)p; bf.u[1] = *(const u32x4*)(p + 16);
    s0 = wmma_bf16(qa[0], bf, s0);
    p = &Ks[ln][32 + 8 * hi];  bf.u[0] = *(const u32x4*)p; bf.u[1] = *(const u32x4*)(p + 16);
    s0 = wmma_bf16(qa[1], bf, s0);
    p = &Ks[16 + ln][8 * hi];      bf.u[0] = *(const u32x4*)p; bf.u[1] = *(const u32x4*)(p + 16);
    s1 = wmma_bf16(qa[0], bf, s1);
    p = &Ks[16 + ln][32 + 8 * hi]; bf.u[0] = *(const u32x4*)p; bf.u[1] = *(const u32x4*)(p + 16);
    s1 = wmma_bf16(qa[1], bf, s1);

    // online softmax (row = r + 8*hi, N striped over 16 lanes of each half)
#pragma unroll
    for (int r = 0; r < 8; ++r) {
      float mv = fmaxf(s0[r], s1[r]);
      mv = fmaxf(mv, __shfl_xor(mv, 1, 32));
      mv = fmaxf(mv, __shfl_xor(mv, 2, 32));
      mv = fmaxf(mv, __shfl_xor(mv, 4, 32));
      mv = fmaxf(mv, __shfl_xor(mv, 8, 32));
      float mnew = fmaxf(mrow[r], mv);
      float al = __expf(mrow[r] - mnew);
      float p0 = __expf(s0[r] - mnew);
      float p1 = __expf(s1[r] - mnew);
      float ps = p0 + p1;
      ps += __shfl_xor(ps, 1, 32);
      ps += __shfl_xor(ps, 2, 32);
      ps += __shfl_xor(ps, 4, 32);
      ps += __shfl_xor(ps, 8, 32);
      lrow[r] = lrow[r] * al + ps;
      mrow[r] = mnew;
      alpha[r] = al;
      Ps[w][(r + 8 * hi) * 40 + ln]      = (__bf16)p0;
      Ps[w][(r + 8 * hi) * 40 + 16 + ln] = (__bf16)p1;
    }
#pragma unroll
    for (int f = 0; f < 4; ++f)
#pragma unroll
      for (int r = 0; r < 8; ++r) co[f][r] *= alpha[r];

    __syncthreads();   // make P (C-layout) visible, reload in A-layout

    AB16 pa;
    const __bf16* pp = &Ps[w][ln * 40 + 8 * hi];
    pa.u[0] = *(const u32x4*)pp;
    pa.u[1] = *(const u32x4*)(pp + 16);
#pragma unroll
    for (int f = 0; f < 4; ++f) {
      const __bf16* vp = &VTs[f * 16 + ln][8 * hi];
      AB16 vb; vb.u[0] = *(const u32x4*)vp; vb.u[1] = *(const u32x4*)(vp + 16);
      co[f] = wmma_bf16(pa, vb, co[f]);
    }
  }

#pragma unroll
  for (int f = 0; f < 4; ++f)
#pragma unroll
    for (int r = 0; r < 8; ++r) {
      int qrow = qbase + r + 8 * hi;
      int col  = f * 16 + ln;
      float val = co[f][r] / lrow[r];
      ctx[((size_t)n * S_ + qrow) * D_ + h * HD_ + col] = (__bf16)val;
    }
}

// ------------------------- residual + LayerNorm ----------------------------
__global__ __launch_bounds__(256)
void add_ln_kernel(const float* __restrict__ a, const float* __restrict__ res,
                   const float* __restrict__ g, const float* __restrict__ b,
                   float* __restrict__ out, __bf16* __restrict__ outb) {
  __shared__ float red[256];
  const int tid = threadIdx.x;
  const size_t base = (size_t)blockIdx.x * D_;
  const int c = tid * 4;
  f32x4 va = *(const f32x4*)(a + base + c);
  f32x4 vr = *(const f32x4*)(res + base + c);
  float x0 = va.x + vr.x, x1 = va.y + vr.y, x2 = va.z + vr.z, x3 = va.w + vr.w;
  red[tid] = x0 + x1 + x2 + x3;
  __syncthreads();
  for (int off = 128; off > 0; off >>= 1) {
    if (tid < off) red[tid] += red[tid + off];
    __syncthreads();
  }
  float mu = red[0] * (1.f / D_);
  __syncthreads();
  float d0 = x0 - mu, d1 = x1 - mu, d2 = x2 - mu, d3 = x3 - mu;
  red[tid] = d0 * d0 + d1 * d1 + d2 * d2 + d3 * d3;
  __syncthreads();
  for (int off = 128; off > 0; off >>= 1) {
    if (tid < off) red[tid] += red[tid + off];
    __syncthreads();
  }
  float rstd = rsqrtf(red[0] * (1.f / D_) + 1e-5f);
  f32x4 gg = *(const f32x4*)(g + c);
  f32x4 bb = *(const f32x4*)(b + c);
  float y0 = d0 * rstd * gg.x + bb.x, y1 = d1 * rstd * gg.y + bb.y;
  float y2 = d2 * rstd * gg.z + bb.z, y3 = d3 * rstd * gg.w + bb.w;
  f32x4 yo = {y0, y1, y2, y3};
  *(f32x4*)(out + base + c) = yo;
  if (outb) {
    outb[base + c + 0] = (__bf16)y0; outb[base + c + 1] = (__bf16)y1;
    outb[base + c + 2] = (__bf16)y2; outb[base + c + 3] = (__bf16)y3;
  }
}

// ------------------------------ orchestration ------------------------------
extern "C" void kernel_launch(void* const* d_in, const int* in_sizes, int n_in,
                              void* d_out, int out_size, void* d_ws, size_t ws_size,
                              hipStream_t stream) {
  const float* src    = (const float*)d_in[0];
  const float* Wq     = (const float*)d_in[1];
  const float* bq     = (const float*)d_in[2];
  const float* Wk     = (const float*)d_in[3];
  const float* bk     = (const float*)d_in[4];
  const float* Wv     = (const float*)d_in[5];
  const float* bv     = (const float*)d_in[6];
  const float* Wo     = (const float*)d_in[7];
  const float* bo     = (const float*)d_in[8];
  const float* W1     = (const float*)d_in[9];
  const float* b1     = (const float*)d_in[10];
  const float* W2     = (const float*)d_in[11];
  const float* b2     = (const float*)d_in[12];
  const float* g_self = (const float*)d_in[13];
  const float* b_self = (const float*)d_in[14];
  const float* g_ffn  = (const float*)d_in[15];
  const float* b_ffn  = (const float*)d_in[16];

  char* ws = (char*)d_ws;
  size_t off = 0;
  auto alloc = [&](size_t bytes) -> void* {
    void* p = ws + off;
    off += (bytes + 255) & ~(size_t)255;
    return p;
  };
  __bf16* srcb = (__bf16*)alloc((size_t)NT_ * D_ * 2);
  __bf16* wqt  = (__bf16*)alloc((size_t)D_ * D_ * 2);
  __bf16* wkt  = (__bf16*)alloc((size_t)D_ * D_ * 2);
  __bf16* wvt  = (__bf16*)alloc((size_t)D_ * D_ * 2);
  __bf16* wot  = (__bf16*)alloc((size_t)D_ * D_ * 2);
  __bf16* w1t  = (__bf16*)alloc((size_t)D_ * FFN_ * 2);
  __bf16* w2t  = (__bf16*)alloc((size_t)FFN_ * D_ * 2);
  __bf16* qh   = (__bf16*)alloc((size_t)NT_ * D_ * 2);
  __bf16* kh   = (__bf16*)alloc((size_t)NT_ * D_ * 2);
  __bf16* vh   = (__bf16*)alloc((size_t)NT_ * D_ * 2);
  __bf16* ctxb = (__bf16*)alloc((size_t)NT_ * D_ * 2);
  float*  tmp  = (float*) alloc((size_t)NT_ * D_ * 4);   // attn_out, later ffn_out
  float*  x    = (float*) alloc((size_t)NT_ * D_ * 4);
  __bf16* xb   = (__bf16*)alloc((size_t)NT_ * D_ * 2);
  __bf16* hbuf = (__bf16*)alloc((size_t)NT_ * FFN_ * 2);

  // 1) precision shuffle: fp32 -> bf16 (+ weight transpose to [out][in])
  f2bf_kernel<<<(NT_ * D_) / (256 * 4), 256, 0, stream>>>(src, srcb);
  wconvert_kernel<<<(D_ * D_) / 256, 256, 0, stream>>>(Wq, wqt, D_, D_);
  wconvert_kernel<<<(D_ * D_) / 256, 256, 0, stream>>>(Wk, wkt, D_, D_);
  wconvert_kernel<<<(D_ * D_) / 256, 256, 0, stream>>>(Wv, wvt, D_, D_);
  wconvert_kernel<<<(D_ * D_) / 256, 256, 0, stream>>>(Wo, wot, D_, D_);
  wconvert_kernel<<<(D_ * FFN_) / 256, 256, 0, stream>>>(W1, w1t, D_, FFN_);
  wconvert_kernel<<<(FFN_ * D_) / 256, 256, 0, stream>>>(W2, w2t, FFN_, D_);

  // 2) QKV projections (1/sqrt(hd)=0.125 folded into q)
  dim3 gDD(D_ / 128, NT_ / 128);
  gemm_bf16_kernel<2><<<gDD, 256, 0, stream>>>(srcb, wqt, bq, qh, NT_, D_, D_, 0.125f);
  gemm_bf16_kernel<2><<<gDD, 256, 0, stream>>>(srcb, wkt, bk, kh, NT_, D_, D_, 1.0f);
  gemm_bf16_kernel<2><<<gDD, 256, 0, stream>>>(srcb, wvt, bv, vh, NT_, D_, D_, 1.0f);

  // 3) flash attention, WMMA scores + WMMA P*V
  attn_kernel<<<dim3(8 * H_, S_ / 128), 256, 0, stream>>>(qh, kh, vh, ctxb);

  // 4) output projection + residual LN1
  gemm_bf16_kernel<0><<<gDD, 256, 0, stream>>>(ctxb, wot, bo, tmp, NT_, D_, D_, 1.0f);
  add_ln_kernel<<<NT_, 256, 0, stream>>>(tmp, src, g_self, b_self, x, xb);

  // 5) FFN (GELU fused in epilogue) + residual LN2 -> d_out
  gemm_bf16_kernel<1><<<dim3(FFN_ / 128, NT_ / 128), 256, 0, stream>>>(
      xb, w1t, b1, hbuf, NT_, FFN_, D_, 1.0f);
  gemm_bf16_kernel<0><<<gDD, 256, 0, stream>>>(hbuf, w2t, b2, tmp, NT_, D_, FFN_, 1.0f);
  add_ln_kernel<<<NT_, 256, 0, stream>>>(tmp, x, g_ffn, b_ffn, (float*)d_out, nullptr);
}
